// BMOJOFAttentionS_16690242912528
// MI455X (gfx1250) — compile-verified, matrix operand order checked by reading, not run on hardware
//
#include <hip/hip_runtime.h>

// ---------- types ----------
typedef __bf16 bh4  __attribute__((ext_vector_type(4)));
typedef __bf16 bh8  __attribute__((ext_vector_type(8)));
typedef __bf16 bh16 __attribute__((ext_vector_type(16)));
typedef float  v8f  __attribute__((ext_vector_type(8)));

union BH16U { bh16 v; __bf16 e[16]; bh8 h[2]; };
union F8U   { v8f  v; float  e[8]; };

// ---------- problem constants ----------
#define Hh   16
#define HKh  4
#define Dh   128
#define Wh   512
#define Gh   (Hh / HKh)      // 4 query heads per kv head
#define KD   (HKh * Dh)      // 512  (K/V row length)
#define QD   (Hh * Dh)       // 2048 (Q/Out row length)
#define BQ   64              // query rows per workgroup
#define BK   32              // keys per tile
#define KST  136             // K LDS row stride (halfs), padded
#define VST  40              // V^T LDS row stride (halfs), padded
#define PST  40              // P LDS row stride (halfs), padded
#define SCALE 0.08838834764831845f   // 1/sqrt(128)

// Process one 32-key tile: S = Q*K^T, online softmax, O += P*V, l += P*1.
// MASKED=false is used for interior tiles that are provably fully in-window.
template<bool MASKED>
__device__ __forceinline__
void swa_tile(int kt, int q0, int wq, int half, int l16, int koff,
              const bh16* __restrict__ qa, F8U* __restrict__ acc /*[9], [8]=row-sum*/,
              float* __restrict__ m, const __bf16* __restrict__ Klds,
              const __bf16* __restrict__ VTlds, __bf16* __restrict__ pw,
              const bh16 ones)
{
    // ---- S = Q * K^T  (two 16x16 tiles over the 32 keys) ----
    F8U c0, c1;
    #pragma unroll
    for (int r = 0; r < 8; ++r) { c0.e[r] = 0.0f; c1.e[r] = 0.0f; }
    #pragma unroll
    for (int c = 0; c < 4; ++c) {
        const __bf16* p0 = &Klds[(l16)      * KST + c * 32 + koff];
        const __bf16* p1 = &Klds[(16 + l16) * KST + c * 32 + koff];
        BH16U b0, b1;
        b0.h[0] = *(const bh8*)(p0);  b0.h[1] = *(const bh8*)(p0 + 16);
        b1.h[0] = *(const bh8*)(p1);  b1.h[1] = *(const bh8*)(p1 + 16);
        c0.v = __builtin_amdgcn_wmma_f32_16x16x32_bf16(false, qa[c], false, b0.v,
                                                       (short)0, c0.v, false, false);
        c1.v = __builtin_amdgcn_wmma_f32_16x16x32_bf16(false, qa[c], false, b1.v,
                                                       (short)0, c1.v, false, false);
    }

    // ---- scale (+mask) + online max + exp ----
    float pr0[8], pr1[8], alp[8];
    const int key0 = kt + l16, key1 = kt + 16 + l16;
    #pragma unroll
    for (int r = 0; r < 8; ++r) {
        float s0, s1;
        if (MASKED) {
            const int qpos = q0 + wq * 16 + half * 8 + r;
            s0 = (key0 <= qpos && key0 + Wh > qpos) ? c0.e[r] * SCALE : -3.0e38f;
            s1 = (key1 <= qpos && key1 + Wh > qpos) ? c1.e[r] * SCALE : -3.0e38f;
        } else {
            s0 = c0.e[r] * SCALE;
            s1 = c1.e[r] * SCALE;
        }
        float mx = fmaxf(s0, s1);
        mx = fmaxf(mx, __shfl_xor(mx, 1));
        mx = fmaxf(mx, __shfl_xor(mx, 2));
        mx = fmaxf(mx, __shfl_xor(mx, 4));
        mx = fmaxf(mx, __shfl_xor(mx, 8));
        float mn = fmaxf(m[r], mx);
        if (MASKED) mn = fmaxf(mn, -1.0e30f);   // all-masked row in this tile -> p == 0
        alp[r] = __expf(m[r] - mn);
        pr0[r] = __expf(s0 - mn);
        pr1[r] = __expf(s1 - mn);
        m[r]   = mn;
    }
    #pragma unroll
    for (int t = 0; t < 9; ++t)
        #pragma unroll
        for (int r = 0; r < 8; ++r) acc[t].e[r] *= alp[r];

    // ---- P: C-layout -> A-layout via per-wave LDS pad ----
    #pragma unroll
    for (int r = 0; r < 8; ++r) {
        const int row = r + half * 8;
        pw[row * PST + l16]      = (__bf16)pr0[r];
        pw[row * PST + 16 + l16] = (__bf16)pr1[r];
    }
    asm volatile("s_wait_dscnt 0" ::: "memory");
    BH16U pa;
    {
        const __bf16* pp = pw + l16 * PST + koff;
        pa.h[0] = *(const bh8*)(pp);
        pa.h[1] = *(const bh8*)(pp + 16);
    }

    // ---- O += P * V (8 tiles over D) ; l += P * 1 (ones-column trick) ----
    #pragma unroll
    for (int t = 0; t < 8; ++t) {
        const __bf16* vp = &VTlds[(t * 16 + l16) * VST + koff];
        BH16U vb;
        vb.h[0] = *(const bh8*)(vp);
        vb.h[1] = *(const bh8*)(vp + 16);
        acc[t].v = __builtin_amdgcn_wmma_f32_16x16x32_bf16(false, pa.v, false, vb.v,
                                                           (short)0, acc[t].v, false, false);
    }
    acc[8].v = __builtin_amdgcn_wmma_f32_16x16x32_bf16(false, pa.v, false, ones,
                                                       (short)0, acc[8].v, false, false);
}

__global__ __launch_bounds__(128)
void swa_fwd_kernel(const float* __restrict__ Q, const float* __restrict__ K,
                    const float* __restrict__ V, float* __restrict__ Out,
                    float* __restrict__ Lse, int T)
{
    __shared__ __align__(16) __bf16 Klds[BK * KST];       //  8.5 KB  [key][dim]
    __shared__ __align__(16) __bf16 VTlds[Dh * VST];      // 10.0 KB  [dim][key]
    __shared__ __align__(16) __bf16 Plds[4][16 * PST];    //  5.0 KB  per-wave P pad

    const int q0   = blockIdx.x * BQ;
    const int h    = blockIdx.y;
    const int hk   = h / Gh;
    const int tid  = threadIdx.x;
    const int wq   = tid >> 5;          // wave id: query strip 16*wq
    const int lane = tid & 31;
    const int half = lane >> 4;         // 0: lanes 0-15, 1: lanes 16-31
    const int l16  = lane & 15;
    const int koff = half * 8;          // A/B fragment K-offset per half-wave

    // ---- Q fragments in WMMA A layout: 4 chunks of 32 dims ----
    bh16 qa[4];
    {
        const int qrow = q0 + wq * 16 + l16;
        const float* qp = Q + (size_t)qrow * QD + h * Dh;
        #pragma unroll
        for (int c = 0; c < 4; ++c) {
            float4 a0 = *(const float4*)(qp + c * 32 + koff);
            float4 a1 = *(const float4*)(qp + c * 32 + koff + 4);
            float4 b0 = *(const float4*)(qp + c * 32 + 16 + koff);
            float4 b1 = *(const float4*)(qp + c * 32 + 16 + koff + 4);
            BH16U t;
            t.e[0]=(__bf16)a0.x;  t.e[1]=(__bf16)a0.y;  t.e[2]=(__bf16)a0.z;  t.e[3]=(__bf16)a0.w;
            t.e[4]=(__bf16)a1.x;  t.e[5]=(__bf16)a1.y;  t.e[6]=(__bf16)a1.z;  t.e[7]=(__bf16)a1.w;
            t.e[8]=(__bf16)b0.x;  t.e[9]=(__bf16)b0.y;  t.e[10]=(__bf16)b0.z; t.e[11]=(__bf16)b0.w;
            t.e[12]=(__bf16)b1.x; t.e[13]=(__bf16)b1.y; t.e[14]=(__bf16)b1.z; t.e[15]=(__bf16)b1.w;
            qa[c] = t.v;
        }
    }

    BH16U onesu;
    #pragma unroll
    for (int i = 0; i < 16; ++i) onesu.e[i] = (__bf16)1.0f;

    // ---- softmax state (rows r + 8*half, replicated over 16 lanes) ----
    float m[8];
    F8U acc[9];                          // [0..7]=O tiles, [8]=row-sum (P*ones)
    #pragma unroll
    for (int r = 0; r < 8; ++r) m[r] = -1.0e30f;
    #pragma unroll
    for (int t = 0; t < 9; ++t)
        #pragma unroll
        for (int r = 0; r < 8; ++r) acc[t].e[r] = 0.0f;

    int kt_lo = q0 - Wh; if (kt_lo < 0) kt_lo = 0;

    for (int kt = kt_lo; kt < q0 + BQ; kt += BK) {
        __syncthreads();   // previous tile fully consumed

        // ---- cooperative stage of K (row-major) and V (transposed) into LDS ----
        for (int i = tid * 4; i < BK * Dh; i += 128 * 4) {
            const int key = i >> 7;
            const int dim = i & 127;
            const size_t gro = (size_t)(kt + key) * KD + hk * Dh + dim;
            float4 kk = *(const float4*)(K + gro);
            bh4 kb; kb[0]=(__bf16)kk.x; kb[1]=(__bf16)kk.y; kb[2]=(__bf16)kk.z; kb[3]=(__bf16)kk.w;
            *(bh4*)(&Klds[key * KST + dim]) = kb;
            float4 vv = *(const float4*)(V + gro);
            VTlds[(dim + 0) * VST + key] = (__bf16)vv.x;
            VTlds[(dim + 1) * VST + key] = (__bf16)vv.y;
            VTlds[(dim + 2) * VST + key] = (__bf16)vv.z;
            VTlds[(dim + 3) * VST + key] = (__bf16)vv.w;
        }
        // prefetch next tile while this one computes
        if (kt + BK < q0 + BQ) {
            const size_t nro = (size_t)(kt + BK + (tid >> 2)) * KD + hk * Dh + (tid & 3) * 32;
            __builtin_prefetch(K + nro, 0, 1);
            __builtin_prefetch(V + nro, 0, 1);
        }
        __syncthreads();

        // interior tiles are fully in-window for every (row, key) pair:
        //   key <= q  for all rows  <=>  kt + BK <= q0 + 1
        //   key >  q-W for all rows <=>  kt >= q0 - Wh + BQ
        const bool full = (kt >= q0 - Wh + BQ) && (kt + BK <= q0 + 1);
        if (full)
            swa_tile<false>(kt, q0, wq, half, l16, koff, qa, acc, m,
                            Klds, VTlds, Plds[wq], onesu.v);
        else
            swa_tile<true>(kt, q0, wq, half, l16, koff, qa, acc, m,
                           Klds, VTlds, Plds[wq], onesu.v);
    }

    // ---- epilogue: O /= l ; lse = m + ln(l) ----
    #pragma unroll
    for (int r = 0; r < 8; ++r) {
        const int qpos = q0 + wq * 16 + half * 8 + r;
        const float l  = acc[8].e[r];
        const float inv = 1.0f / l;
        float* op = Out + (size_t)qpos * QD + h * Dh;
        #pragma unroll
        for (int t = 0; t < 8; ++t) op[t * 16 + l16] = acc[t].e[r] * inv;
        if (l16 == 0) Lse[(size_t)h * T + qpos] = m[r] + __logf(l);
    }
}

extern "C" void kernel_launch(void* const* d_in, const int* in_sizes, int n_in,
                              void* d_out, int out_size, void* d_ws, size_t ws_size,
                              hipStream_t stream) {
    const float* Q = (const float*)d_in[0];
    const float* K = (const float*)d_in[1];
    const float* V = (const float*)d_in[2];
    const int T = in_sizes[0] / QD;                  // 8192
    float* Out = (float*)d_out;                      // [T, H*D]
    float* Lse = (float*)d_out + (size_t)T * QD;     // [H, T]
    dim3 grid(T / BQ, Hh);
    swa_fwd_kernel<<<grid, 128, 0, stream>>>(Q, K, V, Out, Lse, T);
}
